// GCN_26783416058161
// MI455X (gfx1250) — matrix operand order repaired and checked by reading, not compiled, not run on hardware
//
#include <hip/hip_runtime.h>
#include <hip/hip_bf16.h>

// ---------------------------------------------------------------------------
// GCN (3x GCNConv + mean-pool + linear) for MI455X / gfx1250.
//
// Roofline: edge scatter dominates (~2.5 GB of gather+atomic traffic across 3
// layers); whole working set (~100 MB) fits in the 192 MB L2, so scatter runs
// at L2 speed. Dense transforms use v_wmma_f32_16x16x32_f16 (f32 accumulate),
// with dinv[src] folded into the GEMM epilogue and dinv[dst]+bias+relu folded
// into the finish pass. Weights pre-packed once into the exact per-lane WMMA
// B-fragment layout so the GEMM inner loop is pure b128 loads + wmma.
// ---------------------------------------------------------------------------

typedef __attribute__((ext_vector_type(16))) _Float16 v16h;
typedef __attribute__((ext_vector_type(8)))  float    v8f;
typedef __attribute__((ext_vector_type(8)))  unsigned int v8u;

#define N_NODES  100000
#define N_EDGES  1600000
#define F_CH     64
#define N_GRAPHS 128
#define OUT_CH   32

// ---------------------------------------------------------------------------
// degree / dinv
// ---------------------------------------------------------------------------
__global__ void k_deg_init(float* deg, int n) {
    int i = blockIdx.x * blockDim.x + threadIdx.x;
    if (i < n) deg[i] = 1.0f;  // self-loop
}

__global__ void k_deg_count(const long long* __restrict__ dst, float* deg, int e) {
    int i = blockIdx.x * blockDim.x + threadIdx.x;
    if (i < e) atomicAdd(&deg[(int)dst[i]], 1.0f);
}

__global__ void k_deg_rsqrt(float* deg, int n) {
    int i = blockIdx.x * blockDim.x + threadIdx.x;
    if (i < n) {
        float d = deg[i];
        deg[i] = d > 0.0f ? rsqrtf(d) : 0.0f;  // deg >= 1 always
    }
}

// ---------------------------------------------------------------------------
// Pack a 64x64 f32 weight (row-major, W[k][n]) into the per-lane WMMA
// B-fragment layout for v_wmma_f32_16x16x32_f16:
//   lane L holds column n = nt*16 + (L&15); half h holds K = kk*32 + (L<16?0:16) + h
//   dword d packs halves (2d, 2d+1).
// Index: Wp[ ((kk*4 + nt)*32 + lane)*8 + d ]   -> 2048 dwords per weight matrix.
// ---------------------------------------------------------------------------
__global__ void k_pack_w(const float* __restrict__ W, unsigned int* __restrict__ Wp) {
    int t = blockIdx.x * blockDim.x + threadIdx.x;
    if (t >= 2048) return;
    int d    = t & 7;
    int lane = (t >> 3) & 31;
    int nt   = (t >> 8) & 3;
    int kk   = (t >> 10) & 1;
    int n    = nt * 16 + (lane & 15);
    int koff = kk * 32 + ((lane >> 4) << 4);
    int k0   = koff + 2 * d;
    union { unsigned int u; _Float16 h[2]; } p;
    p.h[0] = (_Float16)W[k0 * 64 + n];
    p.h[1] = (_Float16)W[(k0 + 1) * 64 + n];
    Wp[t] = p.u;
}

// ---------------------------------------------------------------------------
// Wave-tile GEMM: g[row][*] = dinv[row] * (A[row][*] @ W), rows in 16-row
// tiles per wave, N=64 as four 16x16 tiles, K=64 as two k-steps of 32.
// Writes the result twice: g (gather source) and agg (accumulator init ==
// self-loop message numerator).
// ---------------------------------------------------------------------------
__global__ void k_gemm_scaled(const float* __restrict__ A,
                              const unsigned int* __restrict__ Wp,
                              const float* __restrict__ dinv,
                              float* __restrict__ g,
                              float* __restrict__ agg,
                              int nrows) {
    const int lane = threadIdx.x & 31;
    const int wave = (blockIdx.x * (blockDim.x >> 5)) + (threadIdx.x >> 5);
    const int m0 = wave * 16;
    if (m0 >= nrows) return;

    const int arow = m0 + (lane & 15);
    const int abase = (lane >> 4) << 4;  // 0 or 16

    // A fragments for both k-steps (contiguous k = abase + s, same order as Wp): two contiguous
    // 8-float runs per lane per k-step.
    v16h a0, a1;
    {
        const float* ar0 = A + arow * 64 + 0 + abase;
        const float* ar1 = A + arow * 64 + 32 + abase;
#pragma unroll
        for (int t = 0; t < 8; ++t) {
            a0[t]     = (_Float16)ar0[t];
            a0[8 + t] = (_Float16)ar0[8 + t];
            a1[t]     = (_Float16)ar1[t];
            a1[8 + t] = (_Float16)ar1[8 + t];
        }
    }

#pragma unroll
    for (int nt = 0; nt < 4; ++nt) {
        v8f c = {};
        // kk = 0
        {
            v8u bu = *(const v8u*)(Wp + (((0 * 4 + nt) * 32 + lane) * 8));
            v16h b = __builtin_bit_cast(v16h, bu);
            c = __builtin_amdgcn_wmma_f32_16x16x32_f16(false, a0, false, b,
                                                       (short)0, c, false, false);
        }
        // kk = 1
        {
            v8u bu = *(const v8u*)(Wp + (((1 * 4 + nt) * 32 + lane) * 8));
            v16h b = __builtin_bit_cast(v16h, bu);
            c = __builtin_amdgcn_wmma_f32_16x16x32_f16(false, a1, false, b,
                                                       (short)0, c, false, false);
        }
        // Epilogue: C/D layout -> VGPR r holds M = r (lanes 0-15) or r+8.
        const int n = nt * 16 + (lane & 15);
        const int mrel = (lane >> 4) << 3;
#pragma unroll
        for (int r = 0; r < 8; ++r) {
            int row = m0 + r + mrel;
            float v = c[r] * dinv[row];
            g[row * 64 + n] = v;
            agg[row * 64 + n] = v;
        }
    }
}

// ---------------------------------------------------------------------------
// Edge scatter: agg[dst] += g[src]. One thread per (edge, 4-channel group):
// b128 gather + 4 f32 atomics (L2-resident).
// ---------------------------------------------------------------------------
__global__ void k_scatter(const long long* __restrict__ src,
                          const long long* __restrict__ dst,
                          const float* __restrict__ g,
                          float* __restrict__ agg,
                          int e) {
    long long gid = (long long)blockIdx.x * blockDim.x + threadIdx.x;
    if (gid >= (long long)e * 16) return;
    int ei = (int)(gid >> 4);
    int c4 = ((int)gid & 15) * 4;
    int s = (int)src[ei];
    int d = (int)dst[ei];
    const float4 v = *(const float4*)(g + (size_t)s * 64 + c4);
    float* p = agg + (size_t)d * 64 + c4;
    atomicAdd(p + 0, v.x);
    atomicAdd(p + 1, v.y);
    atomicAdd(p + 2, v.z);
    atomicAdd(p + 3, v.w);
}

// ---------------------------------------------------------------------------
// Finish: h[i][c] = relu(dinv[i]*agg[i][c] + b[c])   (float4 vectorized)
// ---------------------------------------------------------------------------
__global__ void k_finish(const float* __restrict__ agg,
                         const float* __restrict__ dinv,
                         const float* __restrict__ b,
                         float* __restrict__ h, int n) {
    int gid = blockIdx.x * blockDim.x + threadIdx.x;
    if (gid >= n * 16) return;
    int i = gid >> 4;
    int c4 = (gid & 15) * 4;
    float di = dinv[i];
    float4 a = *(const float4*)(agg + (size_t)i * 64 + c4);
    float4 bb = *(const float4*)(b + c4);
    float4 o;
    o.x = fmaxf(di * a.x + bb.x, 0.0f);
    o.y = fmaxf(di * a.y + bb.y, 0.0f);
    o.z = fmaxf(di * a.z + bb.z, 0.0f);
    o.w = fmaxf(di * a.w + bb.w, 0.0f);
    *(float4*)(h + (size_t)i * 64 + c4) = o;
}

// ---------------------------------------------------------------------------
// Mean pool + final linear
// ---------------------------------------------------------------------------
__global__ void k_pool_zero(float* sums, float* cnt) {
    int i = blockIdx.x * blockDim.x + threadIdx.x;
    if (i < N_GRAPHS * F_CH) sums[i] = 0.0f;
    if (i < N_GRAPHS) cnt[i] = 0.0f;
}

__global__ void k_pool_cnt(const long long* __restrict__ batch, float* cnt, int n) {
    int i = blockIdx.x * blockDim.x + threadIdx.x;
    if (i < n) atomicAdd(&cnt[(int)batch[i]], 1.0f);
}

__global__ void k_pool_sum(const long long* __restrict__ batch,
                           const float* __restrict__ h, float* sums, int n) {
    int gid = blockIdx.x * blockDim.x + threadIdx.x;
    if (gid >= n * 64) return;
    int i = gid >> 6;
    int c = gid & 63;
    atomicAdd(&sums[(int)batch[i] * 64 + c], h[gid]);
}

__global__ void k_final(const float* __restrict__ sums,
                        const float* __restrict__ cnt,
                        const float* __restrict__ Wl,
                        const float* __restrict__ bl,
                        float* __restrict__ out) {
    int t = blockIdx.x * blockDim.x + threadIdx.x;
    if (t >= N_GRAPHS * OUT_CH) return;
    int gidx = t >> 5;        // graph
    int o = t & 31;           // output channel
    float inv = 1.0f / fmaxf(cnt[gidx], 1.0f);
    float acc = bl[o];
#pragma unroll 8
    for (int c = 0; c < 64; ++c)
        acc += (sums[gidx * 64 + c] * inv) * Wl[c * 32 + o];
    out[gidx * 32 + o] = acc;
}

// ---------------------------------------------------------------------------
// Driver
// ---------------------------------------------------------------------------
extern "C" void kernel_launch(void* const* d_in, const int* in_sizes, int n_in,
                              void* d_out, int out_size, void* d_ws, size_t ws_size,
                              hipStream_t stream) {
    const float*     x     = (const float*)d_in[0];
    const long long* ei    = (const long long*)d_in[1];   // [2][E] int64
    const long long* batch = (const long long*)d_in[2];   // [N] int64
    const float* W1 = (const float*)d_in[3];  const float* b1 = (const float*)d_in[4];
    const float* W2 = (const float*)d_in[5];  const float* b2 = (const float*)d_in[6];
    const float* W3 = (const float*)d_in[7];  const float* b3 = (const float*)d_in[8];
    const float* Wl = (const float*)d_in[9];  const float* bl = (const float*)d_in[10];
    float* out = (float*)d_out;

    const int N = in_sizes[0] / F_CH;      // 100000
    const int E = in_sizes[1] / 2;         // 1600000
    const long long* src = ei;
    const long long* dst = ei + E;

    // Workspace layout (~77.2 MB)
    float* B0   = (float*)d_ws;            // N*64
    float* B1   = B0 + (size_t)N * 64;     // N*64
    float* B2   = B1 + (size_t)N * 64;     // N*64 (agg)
    float* dinv = B2 + (size_t)N * 64;     // N
    unsigned int* Wp = (unsigned int*)(dinv + N);  // 3 * 2048 dwords
    float* sums = (float*)(Wp + 3 * 2048); // 128*64
    float* cnt  = sums + N_GRAPHS * F_CH;  // 128

    const int T = 256;
    auto cdiv = [](long long a, long long b) { return (int)((a + b - 1) / b); };

    // degree -> dinv (reused by every layer)
    k_deg_init <<<cdiv(N, T), T, 0, stream>>>(dinv, N);
    k_deg_count<<<cdiv(E, T), T, 0, stream>>>(dst, dinv, E);
    k_deg_rsqrt<<<cdiv(N, T), T, 0, stream>>>(dinv, N);

    // pack weights into WMMA B-fragment layout (f16)
    k_pack_w<<<8, T, 0, stream>>>(W1, Wp + 0);
    k_pack_w<<<8, T, 0, stream>>>(W2, Wp + 2048);
    k_pack_w<<<8, T, 0, stream>>>(W3, Wp + 4096);

    const int gemm_grid = cdiv(N, 16 * 8);        // 8 waves/block, 16 rows/wave
    const int scat_grid = cdiv((long long)E * 16, T);
    const int fin_grid  = cdiv((long long)N * 16, T);

    // Layer 1: x -> B0 (g), B2 (agg);   h1 -> B0
    k_gemm_scaled<<<gemm_grid, T, 0, stream>>>(x, Wp + 0, dinv, B0, B2, N);
    k_scatter    <<<scat_grid, T, 0, stream>>>(src, dst, B0, B2, E);
    k_finish     <<<fin_grid,  T, 0, stream>>>(B2, dinv, b1, B0, N);

    // Layer 2: B0 -> B1 (g), B2 (agg);  h2 -> B1
    k_gemm_scaled<<<gemm_grid, T, 0, stream>>>(B0, Wp + 2048, dinv, B1, B2, N);
    k_scatter    <<<scat_grid, T, 0, stream>>>(src, dst, B1, B2, E);
    k_finish     <<<fin_grid,  T, 0, stream>>>(B2, dinv, b2, B1, N);

    // Layer 3: B1 -> B0 (g), B2 (agg);  h3 -> B0
    k_gemm_scaled<<<gemm_grid, T, 0, stream>>>(B1, Wp + 4096, dinv, B0, B2, N);
    k_scatter    <<<scat_grid, T, 0, stream>>>(src, dst, B0, B2, E);
    k_finish     <<<fin_grid,  T, 0, stream>>>(B2, dinv, b3, B0, N);

    // Mean pool + final linear
    k_pool_zero<<<cdiv(N_GRAPHS * F_CH, T), T, 0, stream>>>(sums, cnt);
    k_pool_cnt <<<cdiv(N, T), T, 0, stream>>>(batch, cnt, N);
    k_pool_sum <<<cdiv((long long)N * 64, T), T, 0, stream>>>(batch, B0, sums, N);
    k_final    <<<cdiv(N_GRAPHS * OUT_CH, T), T, 0, stream>>>(sums, cnt, Wl, bl, out);
}
